// LRULayer_20598663151619
// MI455X (gfx1250) — matrix operand order repaired and checked by reading, not compiled, not run on hardware
//
#include <hip/hip_runtime.h>

// ---------------------------------------------------------------------------
// LRU layer for MI455X (gfx1250, wave32, WMMA).
// LN -> in-proj GEMM (split gate/v) -> conv1d as 5 shifted GEMMs ->
// B-proj GEMM (re+im) -> chunked affine scan -> C-proj GEMM + GELU gate ->
// out GEMM + bias + residual.
//
// All GEMMs: bf16 x bf16 -> f32 via v_wmma_f32_16x16x32_bf16.
// B (weight) operands are pre-swizzled into per-lane fragment layout by the
// prep kernels, so the GEMM inner loop is pure b128 global loads + WMMA:
// no LDS staging, no scalar gathers. Weights are L2-resident (<=0.5 MB each).
// ---------------------------------------------------------------------------

constexpr int BB = 4;            // batch
constexpr int TT = 8192;         // time (power of two)
constexpr int DD = 512;          // model dim
constexpr int SS = 256;          // state dim
constexpr int KK = 5;            // conv kernel
constexpr int MM = BB * TT;      // 32768 rows
constexpr int CH = 512;          // scan chunk length
constexpr int NCH = TT / CH;     // 16 chunks per sequence

typedef __attribute__((ext_vector_type(16))) __bf16 bf16x16;
typedef __attribute__((ext_vector_type(8)))  float  floatx8;

__device__ __forceinline__ floatx8 wmma_bf16(bf16x16 a, bf16x16 b, floatx8 c) {
    return __builtin_amdgcn_wmma_f32_16x16x32_bf16(false, a, false, b, (short)0, c,
                                                   false, false);
}

// ---------------------------------------------------------------------------
// Fragment layouts (CDNA5 ISA 7.12.2):
// A 16x32 bf16, lane l: row M = l&15; K-half = (l>>4): elements K = kh+0..7
//   and kh+16..23 with kh = (l>>4)*8 -> two 16B chunks, 32B apart.
// B 32x16 bf16, lane l: col N = l&15; K = (j>>3)*16 + (l>>4)*8 + (j&7), j = 0..15 -> after
//   prep-swizzle this is 16 contiguous bf16 (32B aligned) per lane.
// C/D 16x16 f32: vgpr i -> M = i + 8*(l>=16), N = l&15.
// ---------------------------------------------------------------------------
__device__ __forceinline__ bf16x16 load_afrag(const __bf16* __restrict__ A,
                                              int row, int kbase, int lda) {
    const uint4* p = reinterpret_cast<const uint4*>(A + (size_t)row * lda + kbase);
    union { uint4 u[2]; bf16x16 v; } t;
    t.u[0] = p[0];   // K = kbase + 0..7
    t.u[1] = p[2];   // K = kbase + 16..23
    return t.v;
}

// Shifted A-fragment load for conv taps; zero at batch edges (TT power of 2).
__device__ __forceinline__ bf16x16 load_afrag_shift(const __bf16* __restrict__ V,
                                                    int m, int kbase, int shift) {
    int t = (m & (TT - 1)) + shift;
    union { uint4 u[2]; bf16x16 v; } r;
    r.u[0] = uint4{0u, 0u, 0u, 0u};
    r.u[1] = uint4{0u, 0u, 0u, 0u};
    if (t >= 0 && t < TT) {
        const uint4* p = reinterpret_cast<const uint4*>(V + (size_t)(m + shift) * SS + kbase);
        r.u[0] = p[0];
        r.u[1] = p[2];
    }
    return r.v;
}

// Swizzled-B element decode: linear index -> (k, n) of the logical [K][N]
// B operand. Layout: [(kb*NB + nb)][lane][j], 16 bf16 per lane.
__device__ __forceinline__ void swz_decode(int idx, int N, int& k, int& n) {
    int j    = idx & 15;
    int lane = (idx >> 4) & 31;
    int blk  = idx >> 9;
    int NB   = N >> 4;
    int kb   = blk / NB;
    int nb   = blk - kb * NB;
    k = kb * 32 + ((j >> 3) << 4) + ((lane >> 4) << 3) + (j & 7);
    n = nb * 16 + (lane & 15);
}

__device__ __forceinline__ float gelu_exact(float x) {
    return 0.5f * x * (1.0f + erff(x * 0.70710678118654752f));
}

// ---------------------------------------------------------------------------
// Stage 0: LayerNorm -> bf16.
// ---------------------------------------------------------------------------
__global__ void k_ln(const float* __restrict__ x, const float* __restrict__ w,
                     const float* __restrict__ b, __bf16* __restrict__ h) {
    int row = blockIdx.x;
    int tid = threadIdx.x;               // 256 threads, 2 elems each
    const float* xr = x + (size_t)row * DD;
    __shared__ float red[256];
    __shared__ float mu_s, rs_s;
    float a0 = xr[tid], a1 = xr[tid + 256];
    red[tid] = a0 + a1;
    __syncthreads();
    for (int off = 128; off > 0; off >>= 1) {
        if (tid < off) red[tid] += red[tid + off];
        __syncthreads();
    }
    if (tid == 0) mu_s = red[0] * (1.0f / DD);
    __syncthreads();
    float mu = mu_s;
    float d0 = a0 - mu, d1 = a1 - mu;
    red[tid] = d0 * d0 + d1 * d1;
    __syncthreads();
    for (int off = 128; off > 0; off >>= 1) {
        if (tid < off) red[tid] += red[tid + off];
        __syncthreads();
    }
    if (tid == 0) rs_s = rsqrtf(red[0] * (1.0f / DD) + 1e-5f);
    __syncthreads();
    float rs = rs_s;
    __bf16* hr = h + (size_t)row * DD;
    hr[tid]       = (__bf16)(d0 * rs * w[tid] + b[tid]);
    hr[tid + 256] = (__bf16)(d1 * rs * w[tid + 256] + b[tid + 256]);
}

// ---------------------------------------------------------------------------
// Weight prep: bf16, transposed AND swizzled into WMMA B-fragment layout.
// ---------------------------------------------------------------------------
__global__ void k_prep_win(const float* __restrict__ W_in, __bf16* __restrict__ winS) {
    int idx = blockIdx.x * blockDim.x + threadIdx.x;   // K=DD, N=2S
    if (idx >= DD * 2 * SS) return;
    int k, n;
    swz_decode(idx, 2 * SS, k, n);
    winS[idx] = (__bf16)W_in[(size_t)n * DD + k];
}

__global__ void k_prep_conv(const float* __restrict__ cw, __bf16* __restrict__ cwS) {
    int idx = blockIdx.x * blockDim.x + threadIdx.x;   // KK blocks of (K=S, N=S)
    if (idx >= KK * SS * SS) return;
    int tap = idx / (SS * SS);
    int rem = idx - tap * (SS * SS);
    int k, n;
    swz_decode(rem, SS, k, n);
    cwS[idx] = (__bf16)cw[((size_t)n * SS + k) * KK + tap];
}

__global__ void k_prep_bn(const float* __restrict__ Bre, const float* __restrict__ Bim,
                          const float* __restrict__ gamma_log,
                          __bf16* __restrict__ bnSr, __bf16* __restrict__ bnSi) {
    int idx = blockIdx.x * blockDim.x + threadIdx.x;   // K=S, N=S
    if (idx >= SS * SS) return;
    int k, n;
    swz_decode(idx, SS, k, n);
    float g = expf(gamma_log[n]);
    bnSr[idx] = (__bf16)(Bre[(size_t)n * SS + k] * g);
    bnSi[idx] = (__bf16)(Bim[(size_t)n * SS + k] * g);
}

__global__ void k_prep_c(const float* __restrict__ Cre, const float* __restrict__ Cim,
                         __bf16* __restrict__ cSr, __bf16* __restrict__ cSmi) {
    int idx = blockIdx.x * blockDim.x + threadIdx.x;   // K=S, N=S
    if (idx >= SS * SS) return;
    int k, n;
    swz_decode(idx, SS, k, n);
    cSr[idx]  = (__bf16)Cre[(size_t)n * SS + k];
    cSmi[idx] = (__bf16)(-Cim[(size_t)n * SS + k]);    // pre-negated: Re = sr@Cr - si@Ci
}

__global__ void k_prep_wout(const float* __restrict__ Wout, __bf16* __restrict__ woutS) {
    int idx = blockIdx.x * blockDim.x + threadIdx.x;   // K=S, N=D
    if (idx >= SS * DD) return;
    int k, n;
    swz_decode(idx, DD, k, n);
    woutS[idx] = (__bf16)Wout[(size_t)n * SS + k];
}

__global__ void k_prep_lam(const float* __restrict__ nu_log, const float* __restrict__ theta_log,
                           float* __restrict__ lam_re, float* __restrict__ lam_im,
                           float* __restrict__ aL_re, float* __restrict__ aL_im) {
    int s = blockIdx.x * blockDim.x + threadIdx.x;
    if (s >= SS) return;
    float nu  = expf(nu_log[s]);
    float mod = expf(-nu);
    float th  = expf(theta_log[s]);
    lam_re[s] = mod * cosf(th);
    lam_im[s] = mod * sinf(th);
    float modL = expf(-(float)CH * nu);   // Lam^CH closed form
    float thL  = (float)CH * th;
    aL_re[s] = modL * cosf(thL);
    aL_im[s] = modL * sinf(thL);
}

// ---------------------------------------------------------------------------
// Stage 1: in-projection. hbf[M,512] @ W -> xg f32 | v bf16. Wave: 32x64 tile.
// ---------------------------------------------------------------------------
__global__ void __launch_bounds__(128)
k_gemm_in(const __bf16* __restrict__ A, const __bf16* __restrict__ Bsw,
          float* __restrict__ xg, __bf16* __restrict__ vbf) {
    int lane = threadIdx.x & 31, wave = threadIdx.x >> 5;
    int m0  = blockIdx.x * 128 + wave * 32;
    int nb0 = blockIdx.y * 4;
    constexpr int NB16 = (2 * SS) / 16;
    const bf16x16* Bv = reinterpret_cast<const bf16x16*>(Bsw);
    int ar = lane & 15, kh = (lane >> 4) << 3;
    floatx8 acc[2][4] = {};
    for (int k0 = 0; k0 < DD; k0 += 32) {
        bf16x16 a0 = load_afrag(A, m0 + ar,      k0 + kh, DD);
        bf16x16 a1 = load_afrag(A, m0 + 16 + ar, k0 + kh, DD);
        if (k0 + 32 < DD)
            __builtin_prefetch(A + (size_t)(m0 + ar) * DD + k0 + 32, 0, 1);
        int kb = k0 >> 5;
#pragma unroll
        for (int g = 0; g < 4; ++g) {
            bf16x16 bf = Bv[((size_t)kb * NB16 + nb0 + g) * 32 + lane];
            acc[0][g] = wmma_bf16(a0, bf, acc[0][g]);
            acc[1][g] = wmma_bf16(a1, bf, acc[1][g]);
        }
    }
    int nlc = lane & 15;
#pragma unroll
    for (int h = 0; h < 2; ++h) {
        int mbase = m0 + h * 16 + ((lane >> 4) << 3);
#pragma unroll
        for (int g = 0; g < 4; ++g) {
            int col = (nb0 + g) * 16 + nlc;
#pragma unroll
            for (int i = 0; i < 8; ++i) {
                int row = mbase + i;
                float v = acc[h][g][i];
                if (col < SS) xg[(size_t)row * SS + col] = v;
                else          vbf[(size_t)row * SS + (col - SS)] = (__bf16)v;
            }
        }
    }
}

// ---------------------------------------------------------------------------
// Stage 2: conv1d as 5 shifted SxS GEMM accumulations.
// ---------------------------------------------------------------------------
__global__ void __launch_bounds__(128)
k_conv(const __bf16* __restrict__ V, const __bf16* __restrict__ CWs,
       const float* __restrict__ cb, __bf16* __restrict__ VC) {
    int lane = threadIdx.x & 31, wave = threadIdx.x >> 5;
    int m0  = blockIdx.x * 128 + wave * 32;
    int nb0 = blockIdx.y * 4;
    constexpr int NB16 = SS / 16;
    int ar = lane & 15, kh = (lane >> 4) << 3;
    floatx8 acc[2][4] = {};
    for (int tap = 0; tap < KK; ++tap) {
        const bf16x16* Bv = reinterpret_cast<const bf16x16*>(CWs + (size_t)tap * SS * SS);
        int shift = tap - 2;
        for (int k0 = 0; k0 < SS; k0 += 32) {
            bf16x16 a0 = load_afrag_shift(V, m0 + ar,      k0 + kh, shift);
            bf16x16 a1 = load_afrag_shift(V, m0 + 16 + ar, k0 + kh, shift);
            int kb = k0 >> 5;
#pragma unroll
            for (int g = 0; g < 4; ++g) {
                bf16x16 bf = Bv[((size_t)kb * NB16 + nb0 + g) * 32 + lane];
                acc[0][g] = wmma_bf16(a0, bf, acc[0][g]);
                acc[1][g] = wmma_bf16(a1, bf, acc[1][g]);
            }
        }
    }
    int nlc = lane & 15;
#pragma unroll
    for (int h = 0; h < 2; ++h) {
        int mbase = m0 + h * 16 + ((lane >> 4) << 3);
#pragma unroll
        for (int g = 0; g < 4; ++g) {
            int col = (nb0 + g) * 16 + nlc;
#pragma unroll
            for (int i = 0; i < 8; ++i) {
                int row = mbase + i;
                VC[(size_t)row * SS + col] = (__bf16)(acc[h][g][i] + cb[col]);
            }
        }
    }
}

// ---------------------------------------------------------------------------
// Stage 3: B-projection (re+im share A). Wave: 16x64 tile, 8 accumulators.
// ---------------------------------------------------------------------------
__global__ void __launch_bounds__(128)
k_bgemm(const __bf16* __restrict__ A, const __bf16* __restrict__ Br,
        const __bf16* __restrict__ Bi, float* __restrict__ Ber, float* __restrict__ Bei) {
    int lane = threadIdx.x & 31, wave = threadIdx.x >> 5;
    int m0  = blockIdx.x * 64 + wave * 16;
    int nb0 = blockIdx.y * 4;
    constexpr int NB16 = SS / 16;
    const bf16x16* Bvr = reinterpret_cast<const bf16x16*>(Br);
    const bf16x16* Bvi = reinterpret_cast<const bf16x16*>(Bi);
    int ar = lane & 15, kh = (lane >> 4) << 3;
    floatx8 accr[4] = {}, acci[4] = {};
    for (int k0 = 0; k0 < SS; k0 += 32) {
        bf16x16 a0 = load_afrag(A, m0 + ar, k0 + kh, SS);
        int kb = k0 >> 5;
#pragma unroll
        for (int g = 0; g < 4; ++g) {
            size_t bidx = ((size_t)kb * NB16 + nb0 + g) * 32 + lane;
            accr[g] = wmma_bf16(a0, Bvr[bidx], accr[g]);
            acci[g] = wmma_bf16(a0, Bvi[bidx], acci[g]);
        }
    }
    int mbase = m0 + ((lane >> 4) << 3);
    int nlc   = lane & 15;
#pragma unroll
    for (int g = 0; g < 4; ++g) {
        int col = (nb0 + g) * 16 + nlc;
#pragma unroll
        for (int i = 0; i < 8; ++i) {
            int row = mbase + i;
            Ber[(size_t)row * SS + col] = accr[g][i];
            Bei[(size_t)row * SS + col] = acci[g][i];
        }
    }
}

// ---------------------------------------------------------------------------
// Stage 4: chunked affine scan, 3 passes (chunk map x -> Lam^CH * x + b).
// ---------------------------------------------------------------------------
__global__ void k_scan_part(const float* __restrict__ Ber, const float* __restrict__ Bei,
                            const float* __restrict__ lam_re, const float* __restrict__ lam_im,
                            float* __restrict__ csum) {
    int gid = blockIdx.x * blockDim.x + threadIdx.x;   // (b*NCH + c)*SS + s
    if (gid >= BB * NCH * SS) return;
    int s = gid % SS;
    int c = (gid / SS) % NCH;
    int b = gid / (SS * NCH);
    float lr = lam_re[s], li = lam_im[s];
    size_t base = ((size_t)b * TT + (size_t)c * CH) * SS + s;
    float sr = 0.f, si = 0.f;
    for (int i = 0; i < CH; ++i) {
        float br = Ber[base], bi = Bei[base];
        float nr = lr * sr - li * si + br;
        float ni = lr * si + li * sr + bi;
        sr = nr; si = ni;
        base += SS;
    }
    csum[(size_t)gid * 2]     = sr;
    csum[(size_t)gid * 2 + 1] = si;
}

__global__ void k_scan_carry(const float* __restrict__ csum,
                             const float* __restrict__ aL_re, const float* __restrict__ aL_im,
                             float* __restrict__ carry) {
    int gid = blockIdx.x * blockDim.x + threadIdx.x;   // b*SS + s
    if (gid >= BB * SS) return;
    int s = gid % SS, b = gid / SS;
    float ar = aL_re[s], ai = aL_im[s];
    float cr = 0.f, ci = 0.f;
    for (int c = 0; c < NCH; ++c) {
        size_t idx = ((size_t)b * NCH + c) * SS + s;
        carry[idx * 2]     = cr;
        carry[idx * 2 + 1] = ci;
        float br = csum[idx * 2], bi = csum[idx * 2 + 1];
        float nr = ar * cr - ai * ci + br;
        float ni = ar * ci + ai * cr + bi;
        cr = nr; ci = ni;
    }
}

__global__ void k_scan_final(const float* __restrict__ Ber, const float* __restrict__ Bei,
                             const float* __restrict__ lam_re, const float* __restrict__ lam_im,
                             const float* __restrict__ carry,
                             __bf16* __restrict__ str, __bf16* __restrict__ sti) {
    int gid = blockIdx.x * blockDim.x + threadIdx.x;
    if (gid >= BB * NCH * SS) return;
    int s = gid % SS;
    int c = (gid / SS) % NCH;
    int b = gid / (SS * NCH);
    float lr = lam_re[s], li = lam_im[s];
    float sr = carry[(size_t)gid * 2], si = carry[(size_t)gid * 2 + 1];
    size_t base = ((size_t)b * TT + (size_t)c * CH) * SS + s;
    for (int i = 0; i < CH; ++i) {
        float br = Ber[base], bi = Bei[base];
        float nr = lr * sr - li * si + br;
        float ni = lr * si + li * sr + bi;
        sr = nr; si = ni;
        str[base] = (__bf16)sr;
        sti[base] = (__bf16)si;
        base += SS;
    }
}

// ---------------------------------------------------------------------------
// Stage 5: C-projection (real part) fused with GELU gating.
// ---------------------------------------------------------------------------
__global__ void __launch_bounds__(128)
k_cgemm(const __bf16* __restrict__ Ar, const __bf16* __restrict__ Ai,
        const __bf16* __restrict__ Br, const __bf16* __restrict__ Bmi,
        const float* __restrict__ xg, __bf16* __restrict__ gout) {
    int lane = threadIdx.x & 31, wave = threadIdx.x >> 5;
    int m0  = blockIdx.x * 128 + wave * 32;
    int nb0 = blockIdx.y * 4;
    constexpr int NB16 = SS / 16;
    const bf16x16* Bvr = reinterpret_cast<const bf16x16*>(Br);
    const bf16x16* Bvi = reinterpret_cast<const bf16x16*>(Bmi);
    int ar = lane & 15, kh = (lane >> 4) << 3;
    floatx8 acc[2][4] = {};
    for (int k0 = 0; k0 < SS; k0 += 32) {
        bf16x16 a0r = load_afrag(Ar, m0 + ar,      k0 + kh, SS);
        bf16x16 a1r = load_afrag(Ar, m0 + 16 + ar, k0 + kh, SS);
        bf16x16 a0i = load_afrag(Ai, m0 + ar,      k0 + kh, SS);
        bf16x16 a1i = load_afrag(Ai, m0 + 16 + ar, k0 + kh, SS);
        int kb = k0 >> 5;
#pragma unroll
        for (int g = 0; g < 4; ++g) {
            size_t bidx = ((size_t)kb * NB16 + nb0 + g) * 32 + lane;
            bf16x16 br = Bvr[bidx];
            bf16x16 bi = Bvi[bidx];
            acc[0][g] = wmma_bf16(a0r, br, acc[0][g]);
            acc[0][g] = wmma_bf16(a0i, bi, acc[0][g]);
            acc[1][g] = wmma_bf16(a1r, br, acc[1][g]);
            acc[1][g] = wmma_bf16(a1i, bi, acc[1][g]);
        }
    }
    int nlc = lane & 15;
#pragma unroll
    for (int h = 0; h < 2; ++h) {
        int mbase = m0 + h * 16 + ((lane >> 4) << 3);
#pragma unroll
        for (int g = 0; g < 4; ++g) {
            int col = (nb0 + g) * 16 + nlc;
#pragma unroll
            for (int i = 0; i < 8; ++i) {
                int row = mbase + i;
                float xv = xg[(size_t)row * SS + col];
                gout[(size_t)row * SS + col] = (__bf16)(gelu_exact(xv) * acc[h][g][i]);
            }
        }
    }
}

// ---------------------------------------------------------------------------
// Stage 6: out-projection fused with bias + residual.
// ---------------------------------------------------------------------------
__global__ void __launch_bounds__(128)
k_gemm_out(const __bf16* __restrict__ A, const __bf16* __restrict__ Bsw,
           const float* __restrict__ b_out, const float* __restrict__ xin,
           float* __restrict__ out) {
    int lane = threadIdx.x & 31, wave = threadIdx.x >> 5;
    int m0  = blockIdx.x * 128 + wave * 32;
    int nb0 = blockIdx.y * 4;
    constexpr int NB16 = DD / 16;
    const bf16x16* Bv = reinterpret_cast<const bf16x16*>(Bsw);
    int ar = lane & 15, kh = (lane >> 4) << 3;
    floatx8 acc[2][4] = {};
    for (int k0 = 0; k0 < SS; k0 += 32) {
        bf16x16 a0 = load_afrag(A, m0 + ar,      k0 + kh, SS);
        bf16x16 a1 = load_afrag(A, m0 + 16 + ar, k0 + kh, SS);
        int kb = k0 >> 5;
#pragma unroll
        for (int g = 0; g < 4; ++g) {
            bf16x16 bf = Bv[((size_t)kb * NB16 + nb0 + g) * 32 + lane];
            acc[0][g] = wmma_bf16(a0, bf, acc[0][g]);
            acc[1][g] = wmma_bf16(a1, bf, acc[1][g]);
        }
    }
    int nlc = lane & 15;
#pragma unroll
    for (int h = 0; h < 2; ++h) {
        int mbase = m0 + h * 16 + ((lane >> 4) << 3);
#pragma unroll
        for (int g = 0; g < 4; ++g) {
            int col = (nb0 + g) * 16 + nlc;
#pragma unroll
            for (int i = 0; i < 8; ++i) {
                int row = mbase + i;
                out[(size_t)row * DD + col] =
                    acc[h][g][i] + b_out[col] + xin[(size_t)row * DD + col];
            }
        }
    }
}

// ---------------------------------------------------------------------------
// Host-side orchestration.
// ---------------------------------------------------------------------------
extern "C" void kernel_launch(void* const* d_in, const int* in_sizes, int n_in,
                              void* d_out, int out_size, void* d_ws, size_t ws_size,
                              hipStream_t stream) {
    (void)in_sizes; (void)n_in; (void)out_size; (void)ws_size;
    const float* x         = (const float*)d_in[0];
    const float* ln_w      = (const float*)d_in[1];
    const float* ln_b      = (const float*)d_in[2];
    const float* W_in      = (const float*)d_in[3];
    const float* conv_w    = (const float*)d_in[4];
    const float* conv_b    = (const float*)d_in[5];
    const float* nu_log    = (const float*)d_in[6];
    const float* theta_log = (const float*)d_in[7];
    const float* gamma_log = (const float*)d_in[8];
    const float* B_re      = (const float*)d_in[9];
    const float* B_im      = (const float*)d_in[10];
    const float* C_re      = (const float*)d_in[11];
    const float* C_im      = (const float*)d_in[12];
    const float* W_out     = (const float*)d_in[13];
    const float* b_out     = (const float*)d_in[14];
    float* out = (float*)d_out;

    char* wp = (char*)d_ws;
    auto alloc = [&](size_t bytes) -> char* {
        char* p = wp;
        wp += (bytes + 255) & ~(size_t)255;
        return p;
    };
    __bf16* hbf   = (__bf16*)alloc((size_t)MM * DD * 2);       // LN output
    __bf16* winS  = (__bf16*)alloc((size_t)DD * 2 * SS * 2);   // swizzled weights
    __bf16* cwS   = (__bf16*)alloc((size_t)KK * SS * SS * 2);
    __bf16* bnSr  = (__bf16*)alloc((size_t)SS * SS * 2);
    __bf16* bnSi  = (__bf16*)alloc((size_t)SS * SS * 2);
    __bf16* cSr   = (__bf16*)alloc((size_t)SS * SS * 2);
    __bf16* cSmi  = (__bf16*)alloc((size_t)SS * SS * 2);
    __bf16* woutS = (__bf16*)alloc((size_t)SS * DD * 2);
    float*  lamr  = (float*)alloc(SS * 4);
    float*  lami  = (float*)alloc(SS * 4);
    float*  aLr   = (float*)alloc(SS * 4);
    float*  aLi   = (float*)alloc(SS * 4);
    float*  xg    = (float*)alloc((size_t)MM * SS * 4);
    __bf16* vbf   = (__bf16*)alloc((size_t)MM * SS * 2);
    __bf16* vcbf  = (__bf16*)alloc((size_t)MM * SS * 2);
    float*  Ber   = (float*)alloc((size_t)MM * SS * 4);
    float*  Bei   = (float*)alloc((size_t)MM * SS * 4);
    float*  csum  = (float*)alloc((size_t)BB * NCH * SS * 2 * 4);
    float*  carry = (float*)alloc((size_t)BB * NCH * SS * 2 * 4);
    __bf16* stre  = (__bf16*)alloc((size_t)MM * SS * 2);
    __bf16* stim  = (__bf16*)alloc((size_t)MM * SS * 2);
    __bf16* gbf   = (__bf16*)alloc((size_t)MM * SS * 2);

    // Weight prep + LN.
    k_prep_win <<<(DD * 2 * SS + 255) / 256, 256, 0, stream>>>(W_in, winS);
    k_prep_conv<<<(KK * SS * SS + 255) / 256, 256, 0, stream>>>(conv_w, cwS);
    k_prep_bn  <<<(SS * SS + 255) / 256, 256, 0, stream>>>(B_re, B_im, gamma_log, bnSr, bnSi);
    k_prep_c   <<<(SS * SS + 255) / 256, 256, 0, stream>>>(C_re, C_im, cSr, cSmi);
    k_prep_wout<<<(SS * DD + 255) / 256, 256, 0, stream>>>(W_out, woutS);
    k_prep_lam <<<1, 256, 0, stream>>>(nu_log, theta_log, lamr, lami, aLr, aLi);
    k_ln       <<<MM, 256, 0, stream>>>(x, ln_w, ln_b, hbf);

    // GEMM stages.
    k_gemm_in<<<dim3(MM / 128, (2 * SS) / 64), 128, 0, stream>>>(hbf, winS, xg, vbf);
    k_conv   <<<dim3(MM / 128, SS / 64), 128, 0, stream>>>(vbf, cwS, conv_b, vcbf);
    k_bgemm  <<<dim3(MM / 64, SS / 64), 128, 0, stream>>>(vcbf, bnSr, bnSi, Ber, Bei);

    // Chunked affine scan.
    int nscan = BB * NCH * SS;
    k_scan_part <<<(nscan + 255) / 256, 256, 0, stream>>>(Ber, Bei, lamr, lami, csum);
    k_scan_carry<<<(BB * SS + 255) / 256, 256, 0, stream>>>(csum, aLr, aLi, carry);
    k_scan_final<<<(nscan + 255) / 256, 256, 0, stream>>>(Ber, Bei, lamr, lami, carry, stre, stim);

    // C-projection + GELU gate, then out-projection + bias + residual.
    k_cgemm   <<<dim3(MM / 128, SS / 64), 128, 0, stream>>>(stre, stim, cSr, cSmi, xg, gbf);
    k_gemm_out<<<dim3(MM / 128, DD / 64), 128, 0, stream>>>(gbf, woutS, b_out, x, out);
}
